// GraphMemoryNetwork_59493886984404
// MI455X (gfx1250) — compile-verified, matrix operand order checked by reading, not run on hardware
//
#include <hip/hip_runtime.h>

// ---------------------------------------------------------------------------
// Types
// ---------------------------------------------------------------------------
typedef __bf16 bf16;
typedef __attribute__((ext_vector_type(16))) __bf16 v16bf;
typedef __attribute__((ext_vector_type(8)))  float  v8f;
typedef __attribute__((ext_vector_type(4)))  unsigned int u32x4;

union BF16Frag { v16bf v; u32x4 q[2]; };

__device__ __forceinline__ unsigned short f2bfbits(float f) {
    unsigned u = __builtin_bit_cast(unsigned, f);
    unsigned r = u + 0x7FFFu + ((u >> 16) & 1u);   // round-to-nearest-even
    return (unsigned short)(r >> 16);
}
__device__ __forceinline__ bf16 f2bf(float f) {
    unsigned short h = f2bfbits(f);
    return __builtin_bit_cast(bf16, h);
}
__device__ __forceinline__ unsigned pack2bf(float a, float b) {
    return (unsigned)f2bfbits(a) | ((unsigned)f2bfbits(b) << 16);
}
__device__ __forceinline__ float bflo(unsigned w) {
    return __builtin_bit_cast(float, w << 16);
}
__device__ __forceinline__ float bfhi(unsigned w) {
    return __builtin_bit_cast(float, w & 0xFFFF0000u);
}
__device__ __forceinline__ float sigmoidf(float x) {
    return 1.0f / (1.0f + __expf(-x));
}

// ---------------------------------------------------------------------------
// WMMA GEMM:  C[M,N] = A[M,K] @ W[N,K]^T (+bias1 +bias2)
// A, W row-major bf16. Wave computes a 16(M) x 64(N) strip: one A fragment
// reused across 4 B tiles per K-step. Epilogue writes fp32 and/or bf16.
// ---------------------------------------------------------------------------
__global__ __launch_bounds__(256) void wmma_gemm_bf16(
    const bf16* __restrict__ A,  int lda,
    const bf16* __restrict__ Bw, int ldb,
    const float* __restrict__ bias1,
    const float* __restrict__ bias2,
    float* __restrict__ outF, int ldof,
    bf16*  __restrict__ outB, int ldob,
    int M, int N, int K)
{
    const int lane = threadIdx.x & 31;
    const int wave = threadIdx.x >> 5;
    const int nStrips = N >> 6;                       // N / 64
    const int totalWaves = (M >> 4) * nStrips;
    const int gw = blockIdx.x * 8 + wave;
    if (gw >= totalWaves) return;
    const int mt = gw / nStrips;
    const int ns = gw % nStrips;

    // 16-bit A/B fragment layout (ISA 7.12.2): lanes 0-15 hold K={0..7,16..23}
    // pairs, lanes 16-31 hold K={8..15,24..31}; K pairs are contiguous bf16.
    const int koff = (lane >> 4) << 3;                // 0 or 8 elements
    const int sub  = lane & 15;
    const bf16* aBase = A + (size_t)(mt * 16 + sub) * lda + koff;

    const int colBase = ns * 64;
    const bf16* bBase0 = Bw + (size_t)(colBase + 0 * 16 + sub) * ldb + koff;
    const bf16* bBase1 = Bw + (size_t)(colBase + 1 * 16 + sub) * ldb + koff;
    const bf16* bBase2 = Bw + (size_t)(colBase + 2 * 16 + sub) * ldb + koff;
    const bf16* bBase3 = Bw + (size_t)(colBase + 3 * 16 + sub) * ldb + koff;

    v8f acc0 = {0.f,0.f,0.f,0.f,0.f,0.f,0.f,0.f};
    v8f acc1 = acc0, acc2 = acc0, acc3 = acc0;

    for (int k = 0; k < K; k += 32) {
        BF16Frag a;
        a.q[0] = *reinterpret_cast<const u32x4*>(aBase + k);
        a.q[1] = *reinterpret_cast<const u32x4*>(aBase + k + 16);
        if (k + 32 < K) __builtin_prefetch(aBase + k + 32, 0, 0);

        BF16Frag b;
        b.q[0] = *reinterpret_cast<const u32x4*>(bBase0 + k);
        b.q[1] = *reinterpret_cast<const u32x4*>(bBase0 + k + 16);
        acc0 = __builtin_amdgcn_wmma_f32_16x16x32_bf16(false, a.v, false, b.v,
                                                       (short)0, acc0, false, false);
        b.q[0] = *reinterpret_cast<const u32x4*>(bBase1 + k);
        b.q[1] = *reinterpret_cast<const u32x4*>(bBase1 + k + 16);
        acc1 = __builtin_amdgcn_wmma_f32_16x16x32_bf16(false, a.v, false, b.v,
                                                       (short)0, acc1, false, false);
        b.q[0] = *reinterpret_cast<const u32x4*>(bBase2 + k);
        b.q[1] = *reinterpret_cast<const u32x4*>(bBase2 + k + 16);
        acc2 = __builtin_amdgcn_wmma_f32_16x16x32_bf16(false, a.v, false, b.v,
                                                       (short)0, acc2, false, false);
        b.q[0] = *reinterpret_cast<const u32x4*>(bBase3 + k);
        b.q[1] = *reinterpret_cast<const u32x4*>(bBase3 + k + 16);
        acc3 = __builtin_amdgcn_wmma_f32_16x16x32_bf16(false, a.v, false, b.v,
                                                       (short)0, acc3, false, false);
    }

    // C/D layout: lane<16 -> M=r (VGPR r), lane>=16 -> M=r+8; N = lane&15.
    const int mOff = (lane >> 4) << 3;
    v8f accs[4] = {acc0, acc1, acc2, acc3};
    #pragma unroll
    for (int s = 0; s < 4; ++s) {
        const int col = colBase + s * 16 + sub;
        float bsum = 0.f;
        if (bias1) bsum += bias1[col];
        if (bias2) bsum += bias2[col];
        #pragma unroll
        for (int r = 0; r < 8; ++r) {
            const int m = mt * 16 + mOff + r;
            const float v = accs[s][r] + bsum;
            if (outF) outF[(size_t)m * ldof + col] = v;
            if (outB) outB[(size_t)m * ldob + col] = f2bf(v);
        }
    }
}

// ---------------------------------------------------------------------------
// fp32 -> bf16 strided convert/copy (builds concat buffers + weight casts)
// ---------------------------------------------------------------------------
__global__ void conv_f32_to_bf16(const float* __restrict__ src, int lds_,
                                 bf16* __restrict__ dst, int ldd,
                                 int rows, int cols)
{
    long i = (long)blockIdx.x * blockDim.x + threadIdx.x;
    long total = (long)rows * cols;
    if (i >= total) return;
    long r = i / cols;
    int  c = (int)(i - r * cols);
    dst[r * (long)ldd + c] = f2bf(src[r * (long)lds_ + c]);
}

// ---------------------------------------------------------------------------
// Edge endpoint gather: pair[n] = concat(nodes[e0[n]], nodes[e1[n]])  (bf16)
// ---------------------------------------------------------------------------
__global__ void gather_pair(const bf16* __restrict__ nodes,
                            const int* __restrict__ edges,
                            bf16* __restrict__ pair, int N)
{
    int idx = blockIdx.x * blockDim.x + threadIdx.x;
    if (idx >= N * 512) return;
    int n = idx >> 9;
    int c = idx & 511;
    int e = edges[n * 2 + (c >> 8)];
    pair[idx] = nodes[(size_t)e * 256 + (c & 255)];
}

// ---------------------------------------------------------------------------
// Sequential LSTM scan, H=256, gates=1024, T steps. One 1024-thread WG
// (32 wave32 waves = full WGP). Thread j owns Whh row j:
//   K in [0,128)   -> 64 packed-bf16x2 VGPRs (persistent across all steps)
//   K in [128,256) -> LDS, 65-dword padded stride (conflict-free banks)
// h lives in LDS fp32 (uniform-address broadcast reads). Two barriers/step.
// ---------------------------------------------------------------------------
__global__ __launch_bounds__(1024, 1) void lstm_scan(
    const float* __restrict__ xg,      // [T, 1024], bih+bhh already folded in
    const float* __restrict__ Whh,     // [1024, 256] fp32
    bf16* __restrict__ outB, int ldob, // hidden states out, bf16
    int T)
{
    extern __shared__ char smem[];
    unsigned* lds_w = (unsigned*)smem;                        // 1024 * 65 dwords
    float*    h_lds = (float*)(smem + 1024 * 65 * 4);         // 256
    float*    g_lds = h_lds + 256;                            // 1024

    const int j = threadIdx.x;
    const float* wrow = Whh + (size_t)j * 256;

    unsigned wreg[64];
    #pragma unroll
    for (int i = 0; i < 64; ++i)
        wreg[i] = pack2bf(wrow[2 * i], wrow[2 * i + 1]);
    for (int i = 0; i < 64; ++i)
        lds_w[j * 65 + i] = pack2bf(wrow[128 + 2 * i], wrow[129 + 2 * i]);
    if (j < 256) h_lds[j] = 0.f;
    float c = 0.f;
    __syncthreads();

    for (int t = 0; t < T; ++t) {
        float acc = xg[(size_t)t * 1024 + j];
        #pragma unroll
        for (int i = 0; i < 64; ++i) {           // K in [0,128): registers
            const unsigned w = wreg[i];
            acc = fmaf(bflo(w), h_lds[2 * i],     acc);
            acc = fmaf(bfhi(w), h_lds[2 * i + 1], acc);
        }
        #pragma unroll 8
        for (int i = 0; i < 64; ++i) {           // K in [128,256): LDS
            const unsigned w = lds_w[j * 65 + i];
            acc = fmaf(bflo(w), h_lds[128 + 2 * i], acc);
            acc = fmaf(bfhi(w), h_lds[129 + 2 * i], acc);
        }
        g_lds[j] = acc;
        __syncthreads();

        if (j < 256) {
            const float gi = sigmoidf(g_lds[j]);
            const float gf = sigmoidf(g_lds[256 + j]);
            const float gn = tanhf(g_lds[512 + j]);
            const float go = sigmoidf(g_lds[768 + j]);
            c = gf * c + gi * gn;
            const float h = go * tanhf(c);
            h_lds[j] = h;
            outB[(size_t)t * ldob + j] = f2bf(h);
        }
        __syncthreads();
    }
}

// ---------------------------------------------------------------------------
// Host launch
// ---------------------------------------------------------------------------
extern "C" void kernel_launch(void* const* d_in, const int* in_sizes, int n_in,
                              void* d_out, int out_size, void* d_ws, size_t ws_size,
                              hipStream_t stream)
{
    (void)in_sizes; (void)n_in; (void)out_size; (void)ws_size;
    constexpr int N = 8192, IN = 256, H = 256, MVD = 128, MEM = 128, OUT = 256;

    const float* inputs   = (const float*)d_in[0];
    const int*   edges    = (const int*)  d_in[1];
    const float* memory   = (const float*)d_in[2];
    const float* Wi       = (const float*)d_in[3];
    const float* bi       = (const float*)d_in[4];
    const float* Wo       = (const float*)d_in[5];
    const float* bo       = (const float*)d_in[6];
    const float* edge_W   = (const float*)d_in[7];
    const float* edge_b   = (const float*)d_in[8];
    const float* node_Wih = (const float*)d_in[9];
    const float* node_Whh = (const float*)d_in[10];
    const float* node_bih = (const float*)d_in[11];
    const float* node_bhh = (const float*)d_in[12];
    const float* Wm       = (const float*)d_in[13];
    const float* bm       = (const float*)d_in[14];
    const float* ctrl_Wih = (const float*)d_in[15];
    const float* ctrl_Whh = (const float*)d_in[16];
    const float* ctrl_bih = (const float*)d_in[17];
    const float* ctrl_bhh = (const float*)d_in[18];

    float* out      = (float*)d_out;            // [8192,256]
    float* out_mem  = out + (size_t)N * OUT;    // [8192,128]

    // ---- carve workspace -------------------------------------------------
    char* ws = (char*)d_ws;
    size_t off = 0;
    auto take = [&](size_t bytes) -> char* {
        char* p = ws + off;
        off = (off + bytes + 255) & ~(size_t)255;
        return p;
    };
    float* xg      = (float*)take((size_t)N * 1024 * 4);
    bf16* pairB    = (bf16*) take((size_t)N * 512 * 2);
    bf16* xcat     = (bf16*) take((size_t)N * 512 * 2);   // [inputs | edge_h]
    bf16* ctrlCat  = (bf16*) take((size_t)N * 640 * 2);   // [inputs|nodes|mem]
    bf16* inB      = (bf16*) take((size_t)N * 256 * 2);
    bf16* nodesB   = (bf16*) take((size_t)N * 256 * 2);
    bf16* hscB     = (bf16*) take((size_t)N * 256 * 2);
    bf16* WiB      = (bf16*) take((size_t)256 * 256 * 2);
    bf16* WoB      = (bf16*) take((size_t)256 * 256 * 2);
    bf16* edgeWB   = (bf16*) take((size_t)2 * 256 * 512 * 2);
    bf16* nodeWihB = (bf16*) take((size_t)2 * 1024 * 512 * 2);
    bf16* ctrlWihB = (bf16*) take((size_t)1024 * 640 * 2);
    bf16* WmB      = (bf16*) take((size_t)128 * 256 * 2);

    auto conv = [&](const float* s, int lds_, bf16* d, int ldd, int rows, int cols) {
        long total = (long)rows * cols;
        int blocks = (int)((total + 255) / 256);
        conv_f32_to_bf16<<<blocks, 256, 0, stream>>>(s, lds_, d, ldd, rows, cols);
    };
    auto gemm = [&](const bf16* A, int lda, const bf16* B, int ldb,
                    const float* b1, const float* b2,
                    float* oF, int ldof, bf16* oB, int ldob,
                    int M, int Nn, int K) {
        int waves = (M / 16) * (Nn / 64);
        int blocks = (waves + 7) / 8;
        wmma_gemm_bf16<<<blocks, 256, 0, stream>>>(A, lda, B, ldb, b1, b2,
                                                   oF, ldof, oB, ldob, M, Nn, K);
    };

    // ---- stage bf16 inputs / weights / concat skeletons ------------------
    conv(inputs, IN,  inB,          IN,  N, IN);
    conv(inputs, IN,  xcat,         512, N, IN);          // xcat[:, 0:256]
    conv(inputs, IN,  ctrlCat,      640, N, IN);          // ctrl[:, 0:256]
    conv(memory, MVD, ctrlCat + 512, 640, N, MVD);        // ctrl[:, 512:640]
    conv(Wi,       IN,  WiB,      IN,  H, IN);
    conv(Wo,       H,   WoB,      H,   OUT, H);
    conv(edge_W,   512, edgeWB,   512, 2 * H, 512);
    conv(node_Wih, 512, nodeWihB, 512, 2 * 4 * H, 512);
    conv(ctrl_Wih, 640, ctrlWihB, 640, 4 * H, 640);
    conv(Wm,       H,   WmB,      H,   MEM, H);

    const size_t scanLds = (size_t)1024 * 65 * 4 + 256 * 4 + 1024 * 4; // ~266 KB

    // ---- nodes = inputs @ Wi.T + bi  (bf16 out) --------------------------
    gemm(inB, IN, WiB, IN, bi, nullptr, nullptr, 0, nodesB, H, N, H, IN);

    // ---- relation loop ---------------------------------------------------
    for (int r = 0; r < 2; ++r) {
        gather_pair<<<(N * 512 + 255) / 256, 256, 0, stream>>>(nodesB, edges, pairB, N);
        // edge_h -> xcat[:, 256:512]
        gemm(pairB, 512, edgeWB + (size_t)r * 256 * 512, 512,
             edge_b + (size_t)r * H, nullptr,
             nullptr, 0, xcat + 256, 512, N, H, 512);
        // xg = xcat @ node_Wih[r].T + bih + bhh
        gemm(xcat, 512, nodeWihB + (size_t)r * 1024 * 512, 512,
             node_bih + (size_t)r * 1024, node_bhh + (size_t)r * 1024,
             xg, 1024, nullptr, 0, N, 1024, 512);
        // scan: r=0 -> nodesB (feeds next gather); r=1 -> ctrlCat[:,256:512]
        bf16* hdst = (r == 0) ? nodesB : (ctrlCat + 256);
        int   hld  = (r == 0) ? 256 : 640;
        lstm_scan<<<1, 1024, scanLds, stream>>>(
            xg, node_Whh + (size_t)r * 1024 * 256, hdst, hld, N);
    }

    // ---- memory_vector = nodes @ Wm.T + bm  (nodes read from ctrlCat) ----
    gemm(ctrlCat + 256, 640, WmB, H, bm, nullptr,
         out_mem, MEM, nullptr, 0, N, MEM, H);

    // ---- controller LSTM -------------------------------------------------
    gemm(ctrlCat, 640, ctrlWihB, 640, ctrl_bih, ctrl_bhh,
         xg, 1024, nullptr, 0, N, 1024, 640);
    lstm_scan<<<1, 1024, scanLds, stream>>>(xg, ctrl_Whh, hscB, 256, N);

    // ---- out = hs @ Wo.T + bo --------------------------------------------
    gemm(hscB, H, WoB, H, bo, nullptr, out, OUT, nullptr, 0, N, OUT, H);
}
